// CoKEModel_27290222199112
// MI455X (gfx1250) — matrix operand order repaired and checked by reading, not compiled, not run on hardware
//
#include <hip/hip_runtime.h>
#include <math.h>

// ---------------- model dimensions ----------------
#define NB    2048      // batch
#define NT    12        // tokens (N)
#define NE    256       // embedding (E)
#define NH    8         // heads
#define NDH   32        // head dim
#define NL    6         // layers
#define NHID  1024      // ffn hidden
#define NVOC  22000
#define NOUT  5389
#define NOUTP 5504      // NOUT padded to multiple of 128 (GEMM BN)
#define NM    (NB*NT)   // 24576 GEMM rows for encoder

// ---------------- WMMA types ----------------
typedef __attribute__((ext_vector_type(16))) __bf16 v16bf;
typedef __attribute__((ext_vector_type(8)))  float  v8f;

union FragU { unsigned int u[8]; v16bf v; };

// ---------------- TDM availability probe ----------------
#ifndef COKE_USE_TDM
#if defined(__has_builtin)
#if __has_builtin(__builtin_amdgcn_tensor_load_to_lds) && __has_builtin(__builtin_amdgcn_s_wait_tensorcnt)
#define COKE_USE_TDM 1
#endif
#endif
#endif
#ifndef COKE_USE_TDM
#define COKE_USE_TDM 0
#endif

#if COKE_USE_TDM
typedef __attribute__((ext_vector_type(4))) unsigned int u32x4;
typedef __attribute__((ext_vector_type(8))) int          i32x8;
typedef __attribute__((ext_vector_type(4))) int          i32x4;

__device__ inline void tdm_load(u32x4 g0, i32x8 g1) {
  const i32x4 z4 = {0, 0, 0, 0};
#if defined(__clang_major__) && __clang_major__ >= 23
  const i32x8 z8 = {0, 0, 0, 0, 0, 0, 0, 0};
  __builtin_amdgcn_tensor_load_to_lds(g0, g1, z4, z4, z8, 0);
#else
  __builtin_amdgcn_tensor_load_to_lds(g0, g1, z4, z4, 0);
#endif
}
#endif

// ---------------- helpers ----------------
__device__ inline unsigned short f2bf(float f) {
  unsigned int u = __float_as_uint(f);
  u += 0x7fffu + ((u >> 16) & 1u);          // round-to-nearest-even
  return (unsigned short)(u >> 16);
}

__device__ inline float gelu_f(float x) {
  float x3 = x * x * x;
  return 0.5f * x * (1.0f + tanhf(0.79788456080286535588f * (x + 0.044715f * x3)));
}

__device__ inline float waveReduceSum(float v) {
  #pragma unroll
  for (int off = 16; off > 0; off >>= 1) v += __shfl_xor(v, off, 32);
  return v;
}
__device__ inline float waveReduceMax(float v) {
  #pragma unroll
  for (int off = 16; off > 0; off >>= 1) v = fmaxf(v, __shfl_xor(v, off, 32));
  return v;
}

__device__ inline float blockReduceSum(float v) {   // blockDim.x == 256
  __shared__ float red[8];
  int lane = threadIdx.x & 31, w = threadIdx.x >> 5;
  v = waveReduceSum(v);
  if (lane == 0) red[w] = v;
  __syncthreads();
  float r = 0.f;
  if (w == 0) {
    r = (lane < 8) ? red[lane] : 0.f;
    r = waveReduceSum(r);
    if (lane == 0) red[0] = r;
  }
  __syncthreads();
  float out = red[0];
  __syncthreads();
  return out;
}
__device__ inline float blockReduceMax(float v) {   // blockDim.x == 256
  __shared__ float redm[8];
  int lane = threadIdx.x & 31, w = threadIdx.x >> 5;
  v = waveReduceMax(v);
  if (lane == 0) redm[w] = v;
  __syncthreads();
  float r = -1e30f;
  if (w == 0) {
    r = (lane < 8) ? redm[lane] : -1e30f;
    r = waveReduceMax(r);
    if (lane == 0) redm[0] = r;
  }
  __syncthreads();
  float out = redm[0];
  __syncthreads();
  return out;
}

// =====================================================================
// GEMM: C[M,Nout] = A[M,K](bf16) * W[K,Npad](bf16, zero padded) + bias
// epilogue: bit0 = gelu, bit1 = bf16 output (else f32)
// BM=64 BN=128 BK=32, 256 threads (8 wave32), wave -> 32x32 = 2x2 WMMA
// Staging: TDM (tensor_load_to_lds, pad-enabled to reproduce the +2-elem
// bank-skewed LDS strides) when available; manual b128+ds fallback else.
// =====================================================================
#define BM 64
#define BN 128
#define BK 32

__global__ __launch_bounds__(256) void gemm_bf16_kernel(
    const unsigned short* __restrict__ A,   // [M,K] bf16, M%64==0, K%32==0
    const unsigned short* __restrict__ W,   // [K,Npad] bf16, Npad%128==0
    const float* __restrict__ bias,         // [Nout] or nullptr
    void* __restrict__ out,                 // [M,Nout] f32 or bf16
    int M, int K, int Npad, int Nout, int epilogue)
{
  __shared__ unsigned short As[BM][BK + 2];   // +2 pad: bank skew (TDM pad matches)
  __shared__ unsigned short Bs[BK][BN + 2];

  const int tid = threadIdx.x;
  const int nBlocks = Npad / BN;
  const int mBase = (blockIdx.x / nBlocks) * BM;
  const int nBase = (blockIdx.x % nBlocks) * BN;

  const int wave = tid >> 5;
  const int lane = tid & 31;
  const int wm = (wave & 1) * 32;      // wave M offset inside tile
  const int wn = (wave >> 1) * 32;     // wave N offset inside tile
  const int lrow = lane & 15;
  const int half = lane >> 4;

  v8f acc[2][2];
  const v8f vzero = {0.f,0.f,0.f,0.f,0.f,0.f,0.f,0.f};
  acc[0][0] = vzero; acc[0][1] = vzero; acc[1][0] = vzero; acc[1][1] = vzero;

#if COKE_USE_TDM
  const unsigned ldsA = (unsigned)(size_t)(void*)&As[0][0];
  const unsigned ldsB = (unsigned)(size_t)(void*)&Bs[0][0];
#else
  // cooperative tile-load mapping (manual fallback)
  const int arow = tid >> 2;            // 0..63
  const int acol = (tid & 3) * 8;       // 0,8,16,24
  const int wrow = tid >> 3;            // 0..31
  const int wcol = (tid & 7) * 16;      // 0..112
#endif

  for (int k0 = 0; k0 < K; k0 += BK) {
#if COKE_USE_TDM
    if (wave == 0) {
      // ---- A tile: 64 rows x 32 cols bf16, row stride K elems ----
      // pad_interval=3 (pad every 16 DWORDs) + 1 DWORD -> LDS stride 34 elems
      unsigned long long ga =
          (unsigned long long)(size_t)(const void*)(A + (size_t)mBase * K + k0);
      u32x4 g0a = { 1u,                                  // count=1 (valid D#)
                    ldsA,                                // lds_addr
                    (unsigned)ga,                        // global_addr[31:0]
                    (unsigned)((ga >> 32) & 0x01FFFFFFu) | (2u << 30) }; // type=2
      i32x8 g1a = { (int)((1u << 16) | (1u << 20) | (3u << 22)), // 2B, pad_en, int=16DW
                    (int)(((unsigned)K & 0xFFFFu) << 16),        // tensor_dim0 lo16
                    (int)(((unsigned)K >> 16)),                  // dim0 hi16 | dim1 lo16
                    (int)(0x10u | (32u << 16)),                  // dim1 hi16 | tile_dim0=32
                    64,                                          // tile_dim1=64
                    (int)K,                                      // tensor_dim0_stride
                    0, 0 };
      tdm_load(g0a, g1a);
      // ---- W tile: 32 rows x 128 cols bf16, row stride Npad elems ----
      // pad_interval=5 (pad every 64 DWORDs) + 1 DWORD -> LDS stride 130 elems
      unsigned long long gw =
          (unsigned long long)(size_t)(const void*)(W + (size_t)k0 * Npad + nBase);
      u32x4 g0w = { 1u, ldsB, (unsigned)gw,
                    (unsigned)((gw >> 32) & 0x01FFFFFFu) | (2u << 30) };
      i32x8 g1w = { (int)((1u << 16) | (1u << 20) | (5u << 22)),
                    (int)(((unsigned)Npad & 0xFFFFu) << 16),
                    (int)(((unsigned)Npad >> 16)),
                    (int)(0x10u | (128u << 16)),                 // tile_dim0=128
                    32,                                          // tile_dim1=32
                    (int)Npad,
                    0, 0 };
      tdm_load(g0w, g1w);
    }
    if (k0 + BK < K) {
      __builtin_prefetch(A + (size_t)(mBase + (tid >> 2)) * K + k0 + BK, 0, 0);
      __builtin_prefetch(W + (size_t)(k0 + BK + (tid >> 3)) * Npad + nBase, 0, 0);
    }
    __builtin_amdgcn_s_wait_tensorcnt(0);
    __syncthreads();
#else
    // ---- stage A tile (64x32 bf16) : one b128 load per thread ----
    {
      const unsigned short* src = A + (size_t)(mBase + arow) * K + k0 + acol;
      uint4 d = *(const uint4*)src;
      unsigned int* dst = (unsigned int*)&As[arow][acol];
      dst[0] = d.x; dst[1] = d.y; dst[2] = d.z; dst[3] = d.w;
    }
    // ---- stage W tile (32x128 bf16) : two b128 loads per thread ----
    {
      const unsigned short* src = W + (size_t)(k0 + wrow) * Npad + nBase + wcol;
      uint4 d0 = *(const uint4*)src;
      uint4 d1 = *(const uint4*)(src + 8);
      unsigned int* dst = (unsigned int*)&Bs[wrow][wcol];
      dst[0] = d0.x; dst[1] = d0.y; dst[2] = d0.z; dst[3] = d0.w;
      dst[4] = d1.x; dst[5] = d1.y; dst[6] = d1.z; dst[7] = d1.w;
    }
    if (k0 + BK < K) {
      __builtin_prefetch(A + (size_t)(mBase + arow) * K + k0 + BK + acol, 0, 0);
      __builtin_prefetch(W + (size_t)(k0 + BK + wrow) * Npad + nBase + wcol, 0, 0);
    }
    __syncthreads();
#endif

    // ---- assemble fragments, contiguous k = 16*half + s for both A and B ----
    FragU fa[2], fb[2];
    #pragma unroll
    for (int s = 0; s < 2; s++) {
      const int mrow = wm + s * 16 + lrow;
      #pragma unroll
      for (int v = 0; v < 8; v++) {
        // A 16x32: lanes0-15 hold K=0..15 (2 per VGPR), lanes16-31 K=16..31
        const int kb = 16 * half + 2 * v;
        fa[s].u[v] = *(const unsigned int*)&As[mrow][kb];
      }
    }
    #pragma unroll
    for (int t = 0; t < 2; t++) {
      const int ncol = wn + t * 16 + lrow;
      #pragma unroll
      for (int v = 0; v < 8; v++) {
        // B 32x16: lanes0-15 hold K=0..15 (2 per VGPR), lanes16-31 K=16..31
        const int kb = 16 * half + 2 * v;
        unsigned int lo = Bs[kb][ncol];
        unsigned int hi = Bs[kb + 1][ncol];
        fb[t].u[v] = lo | (hi << 16);
      }
    }
    #pragma unroll
    for (int s = 0; s < 2; s++)
      #pragma unroll
      for (int t = 0; t < 2; t++)
        acc[s][t] = __builtin_amdgcn_wmma_f32_16x16x32_bf16(
            false, fa[s].v, false, fb[t].v, (short)0, acc[s][t], false, false);
    __syncthreads();
  }

  // ---- epilogue ----
  #pragma unroll
  for (int s = 0; s < 2; s++) {
    const int mTile = mBase + wm + s * 16;
    #pragma unroll
    for (int t = 0; t < 2; t++) {
      const int col = nBase + wn + t * 16 + lrow;
      if (col >= Nout) continue;
      const float bv = bias ? bias[col] : 0.0f;
      #pragma unroll
      for (int r = 0; r < 8; r++) {
        const int row = mTile + r + 8 * half;   // C/D: VGPR r -> M = r + 8*half
        float vacc = acc[s][t][r] + bv;
        if (epilogue & 1) vacc = gelu_f(vacc);
        if (epilogue & 2)
          ((unsigned short*)out)[(size_t)row * Nout + col] = f2bf(vacc);
        else
          ((float*)out)[(size_t)row * Nout + col] = vacc;
      }
    }
  }
}

// =====================================================================
// f32 -> bf16 conversion with column padding (for GEMM weights)
// =====================================================================
__global__ void convert_pad_kernel(const float* __restrict__ src,
                                   unsigned short* __restrict__ dst,
                                   long rows, long cols, long dstStride)
{
  long n = rows * dstStride;
  for (long i = (long)blockIdx.x * blockDim.x + threadIdx.x; i < n;
       i += (long)gridDim.x * blockDim.x) {
    long r = i / dstStride, c = i % dstStride;
    dst[i] = (c < cols) ? f2bf(src[r * cols + c]) : (unsigned short)0;
  }
}

// =====================================================================
// x0 = LN(ele_emb[src])  -- one block per token row, E=256
// =====================================================================
__global__ __launch_bounds__(256) void embed_ln_kernel(
    const int* __restrict__ src, const float* __restrict__ emb,
    float* __restrict__ xf, unsigned short* __restrict__ xb)
{
  const int row = blockIdx.x;
  const int e = threadIdx.x;
  const int tok = src[row];
  float v = emb[(size_t)tok * NE + e];
  float mean = blockReduceSum(v) * (1.0f / NE);
  float d = v - mean;
  float var = blockReduceSum(d * d) * (1.0f / NE);
  float o = d * rsqrtf(var + 1e-12f);
  size_t idx = (size_t)row * NE + e;
  xf[idx] = o;
  xb[idx] = f2bf(o);
}

// =====================================================================
// spatial+relation bias build (3 mask variants), one thread per (b,h,i)
// =====================================================================
__global__ void bias_build_kernel(
    const int* __restrict__ dist, const int* __restrict__ rel,
    const float* __restrict__ mask,
    const float* __restrict__ spatial_emb, const float* __restrict__ virt,
    const float* __restrict__ rel_emb, const float* __restrict__ rel_w,
    float* __restrict__ bias0, float* __restrict__ bias1, float* __restrict__ bias2)
{
  int gid = blockIdx.x * blockDim.x + threadIdx.x;
  if (gid >= NB * NH * NT) return;
  const int b = gid / (NH * NT);
  const int rem = gid % (NH * NT);
  const int h = rem / NT;
  const int i = rem % NT;

  float s[NT], r[NT];
  const float tv = virt[h];
  // spatial bias row + graph-virt slice adds
  for (int j = 0; j < NT; j++) {
    int d = dist[((size_t)b * NT + i) * NT + j];
    float v = spatial_emb[d * NH + h];
    if (i == 0) v += tv;            // spb[:, :, 0, :] += t
    else if (j == 0) v += tv;       // spb[:, :, 1:, 0] += t
    s[j] = v;
  }
  { // LN over j
    float m = 0; for (int j = 0; j < NT; j++) m += s[j];
    m *= (1.0f / NT);
    float var = 0; for (int j = 0; j < NT; j++) { float dd = s[j] - m; var += dd * dd; }
    var *= (1.0f / NT);
    float inv = rsqrtf(var + 1e-12f);
    for (int j = 0; j < NT; j++) s[j] = (s[j] - m) * inv;
  }
  // relation bias row: rb[b,h,i,j] = sum_k rel_emb[rel][k] * rw[b,k,h]
  for (int j = 0; j < NT; j++) {
    int ri = rel[((size_t)b * NT + i) * NT + j];
    float acc = 0.f;
    for (int k = 0; k < NH; k++)
      acc += rel_emb[ri * NH + k] * rel_w[(size_t)b * NH * NH + k * NH + h];
    r[j] = acc;
  }
  { // LN over j
    float m = 0; for (int j = 0; j < NT; j++) m += r[j];
    m *= (1.0f / NT);
    float var = 0; for (int j = 0; j < NT; j++) { float dd = r[j] - m; var += dd * dd; }
    var *= (1.0f / NT);
    float inv = rsqrtf(var + 1e-12f);
    for (int j = 0; j < NT; j++) r[j] = (r[j] - m) * inv;
  }
  // masks + write all three bias tensors
  for (int j = 0; j < NT; j++) {
    float mv = mask[((size_t)b * NT + i) * NT + j];
    size_t o = (((size_t)b * NH + h) * NT + i) * NT + j;
    float base = s[j] + r[j];
    bias0[o] = base + (1.0f - mv) * -1e9f;
    bool z1 = (i >= 1 && i < 4) || (i >= 6 && i < 11) || (j >= 1 && j < 4) || (j >= 6 && j < 11);
    bias1[o] = base + (1.0f - (z1 ? 0.f : mv)) * -1e9f;
    bool z2 = (i >= 1 && i < 6) || (i >= 8 && i < 11) || (j >= 1 && j < 6) || (j >= 8 && j < 11);
    bias2[o] = base + (1.0f - (z2 ? 0.f : mv)) * -1e9f;
  }
}

// =====================================================================
// attention: one wave32 per (b, i, h); lane = head-dim element
// =====================================================================
__global__ __launch_bounds__(256) void attn_kernel(
    const float* __restrict__ qf, const float* __restrict__ kf,
    const float* __restrict__ vf, const float* __restrict__ bias,
    unsigned short* __restrict__ ctxb)
{
  const int wid = (blockIdx.x << 3) + (threadIdx.x >> 5);
  const int lane = threadIdx.x & 31;
  const int b = wid / (NT * NH);
  const int rem = wid % (NT * NH);
  const int i = rem / NH;
  const int h = rem % NH;
  const float scale = 0.17677669529663687f;   // 1/sqrt(32)

  const float qd = qf[((size_t)(b * NT + i)) * NE + h * NDH + lane];
  float sc[NT];
  float mx = -1e30f;
  #pragma unroll
  for (int j = 0; j < NT; j++) {
    float kd = kf[((size_t)(b * NT + j)) * NE + h * NDH + lane];
    float s = waveReduceSum(qd * kd) * scale
            + bias[(((size_t)b * NH + h) * NT + i) * NT + j];
    sc[j] = s;
    mx = fmaxf(mx, s);
  }
  float se = 0.f;
  #pragma unroll
  for (int j = 0; j < NT; j++) { sc[j] = expf(sc[j] - mx); se += sc[j]; }
  const float inv = 1.0f / se;
  float acc = 0.f;
  #pragma unroll
  for (int j = 0; j < NT; j++)
    acc += sc[j] * vf[((size_t)(b * NT + j)) * NE + h * NDH + lane];
  ctxb[((size_t)(b * NT + i)) * NE + h * NDH + lane] = f2bf(acc * inv);
}

// =====================================================================
// x = LN(xin + y) * gamma + beta  -> f32 and bf16 copies
// =====================================================================
__global__ __launch_bounds__(256) void residual_ln_kernel(
    const float* __restrict__ xin, const float* __restrict__ y,
    const float* __restrict__ gamma, const float* __restrict__ beta,
    float* __restrict__ xoutf, unsigned short* __restrict__ xoutb)
{
  const int row = blockIdx.x;
  const int e = threadIdx.x;
  const size_t idx = (size_t)row * NE + e;
  float v = xin[idx] + y[idx];
  float mean = blockReduceSum(v) * (1.0f / NE);
  float d = v - mean;
  float var = blockReduceSum(d * d) * (1.0f / NE);
  float o = d * rsqrtf(var + 1e-12f) * gamma[e] + beta[e];
  xoutf[idx] = o;
  xoutb[idx] = f2bf(o);
}

// =====================================================================
// gather one token row per batch element into head-input buffer (bf16)
// =====================================================================
__global__ void gather_rows_kernel(const unsigned short* __restrict__ xb,
                                   unsigned short* __restrict__ sel, int tok)
{
  int idx = blockIdx.x * blockDim.x + threadIdx.x;
  if (idx >= NB * NE) return;
  int b = idx / NE, e = idx % NE;
  sel[idx] = xb[((size_t)(b * NT + tok)) * NE + e];
}

// =====================================================================
// head mid: t = LN(gelu(g1)) * post_scale + post_bias  (per-batch params)
// =====================================================================
__global__ __launch_bounds__(256) void head_mid_kernel(
    const float* __restrict__ g1, const float* __restrict__ ps,
    const float* __restrict__ pb, unsigned short* __restrict__ tb)
{
  const int b = blockIdx.x;
  const int e = threadIdx.x;
  const size_t idx = (size_t)b * NE + e;
  float g = gelu_f(g1[idx]);
  float mean = blockReduceSum(g) * (1.0f / NE);
  float d = g - mean;
  float var = blockReduceSum(d * d) * (1.0f / NE);
  float o = d * rsqrtf(var + 1e-12f);
  tb[idx] = f2bf(o * ps[idx] + pb[idx]);
}

// =====================================================================
// label-smoothing loss per row: -conf * log_softmax(lg)[target] / B
// =====================================================================
__global__ __launch_bounds__(256) void loss_row_kernel(
    const float* __restrict__ lg, const int* __restrict__ tgt,
    float* __restrict__ rowloss)
{
  const int b = blockIdx.x;
  const float* row = lg + (size_t)b * NOUT;
  float mx = -1e30f;
  for (int c = threadIdx.x; c < NOUT; c += 256) mx = fmaxf(mx, row[c]);
  mx = blockReduceMax(mx);
  float se = 0.f;
  for (int c = threadIdx.x; c < NOUT; c += 256) se += expf(row[c] - mx);
  se = blockReduceSum(se);
  if (threadIdx.x == 0) {
    int t = tgt[b];
    rowloss[b] = -0.9f * (row[t] - mx - logf(se)) * (1.0f / NB);
  }
}

__global__ __launch_bounds__(256) void loss_final_kernel(
    const float* __restrict__ r1, const float* __restrict__ r2,
    float* __restrict__ out)
{
  float s = 0.f;
  for (int i = threadIdx.x; i < NB; i += 256) s += r1[i] + r2[i];
  s = blockReduceSum(s);
  if (threadIdx.x == 0) out[0] = s;
}

// =====================================================================
// host orchestration
// =====================================================================
static void launch_gemm(hipStream_t st, const unsigned short* A,
                        const unsigned short* W, const float* bias, void* out,
                        int M, int K, int Npad, int Nout, int ep)
{
  dim3 grid((M / BM) * (Npad / BN));
  gemm_bf16_kernel<<<grid, 256, 0, st>>>(A, W, bias, out, M, K, Npad, Nout, ep);
}

extern "C" void kernel_launch(void* const* d_in, const int* in_sizes, int n_in,
                              void* d_out, int out_size, void* d_ws, size_t ws_size,
                              hipStream_t stream)
{
  (void)in_sizes; (void)n_in; (void)out_size; (void)ws_size;
  // ---- inputs (setup_inputs dict order, params flattened in insertion order)
  const int*   src        = (const int*)d_in[0];
  const int*   target     = (const int*)d_in[1];
  const int*   dist       = (const int*)d_in[2];
  const int*   rel        = (const int*)d_in[3];
  const float* mask       = (const float*)d_in[4];
  const float* ele_emb    = (const float*)d_in[5];
  const float* sp_emb     = (const float*)d_in[6];
  const float* graph_virt = (const float*)d_in[7];
  const float* rel_emb    = (const float*)d_in[8];
  const float* rel_w      = (const float*)d_in[9];
  const float* post_scale = (const float*)d_in[10];
  const float* post_bias  = (const float*)d_in[11];
  const float* fc1_w = (const float*)d_in[12]; const float* fc1_b = (const float*)d_in[13];
  const float* fc2_w = (const float*)d_in[14]; const float* fc2_b = (const float*)d_in[15];
  const float* wq = (const float*)d_in[16]; const float* bq = (const float*)d_in[17];
  const float* wk = (const float*)d_in[18]; const float* bk = (const float*)d_in[19];
  const float* wv = (const float*)d_in[20]; const float* bv = (const float*)d_in[21];
  const float* wo = (const float*)d_in[22]; const float* bo = (const float*)d_in[23];
  const float* ln1g = (const float*)d_in[24]; const float* ln1b = (const float*)d_in[25];
  const float* w1 = (const float*)d_in[26]; const float* b1 = (const float*)d_in[27];
  const float* w2 = (const float*)d_in[28]; const float* b2 = (const float*)d_in[29];
  const float* ln2g = (const float*)d_in[30]; const float* ln2b = (const float*)d_in[31];

  // ---- workspace carve-out (256B aligned) ----
  char* ws = (char*)d_ws;
  size_t off = 0;
  auto alloc = [&](size_t bytes) -> void* {
    void* p = ws + off;
    off += (bytes + 255) & ~(size_t)255;
    return p;
  };
  unsigned short* wqb  = (unsigned short*)alloc((size_t)NL * NE * NE * 2);
  unsigned short* wkb  = (unsigned short*)alloc((size_t)NL * NE * NE * 2);
  unsigned short* wvb  = (unsigned short*)alloc((size_t)NL * NE * NE * 2);
  unsigned short* wob  = (unsigned short*)alloc((size_t)NL * NE * NE * 2);
  unsigned short* w1b  = (unsigned short*)alloc((size_t)NL * NE * NHID * 2);
  unsigned short* w2b  = (unsigned short*)alloc((size_t)NL * NHID * NE * 2);
  unsigned short* fc1b = (unsigned short*)alloc((size_t)NE * NE * 2);
  unsigned short* fc2b = (unsigned short*)alloc((size_t)NE * NOUTP * 2);
  float*          x0f  = (float*)alloc((size_t)NM * NE * 4);
  unsigned short* x0b  = (unsigned short*)alloc((size_t)NM * NE * 2);
  float*          xf   = (float*)alloc((size_t)NM * NE * 4);
  unsigned short* xb   = (unsigned short*)alloc((size_t)NM * NE * 2);
  float*          qf   = (float*)alloc((size_t)NM * NE * 4);   // also GEMM tmp
  float*          kf   = (float*)alloc((size_t)NM * NE * 4);
  float*          vf   = (float*)alloc((size_t)NM * NE * 4);
  unsigned short* ctxb = (unsigned short*)alloc((size_t)NM * NE * 2);
  unsigned short* h1b  = (unsigned short*)alloc((size_t)NM * NHID * 2);
  float*          bias0 = (float*)alloc((size_t)NB * NH * NT * NT * 4);
  float*          bias1 = (float*)alloc((size_t)NB * NH * NT * NT * 4);
  float*          bias2 = (float*)alloc((size_t)NB * NH * NT * NT * 4);
  unsigned short* selb = (unsigned short*)alloc((size_t)NB * NE * 2);
  float*          g1f  = (float*)alloc((size_t)NB * NE * 4);
  unsigned short* tb   = (unsigned short*)alloc((size_t)NB * NE * 2);
  float*          lgt  = (float*)alloc((size_t)NB * NOUT * 4);
  float*          lb1  = (float*)alloc((size_t)NB * 4);
  float*          lb2  = (float*)alloc((size_t)NB * 4);

  // ---- weight conversions to bf16 (fc2 column-padded to NOUTP) ----
  convert_pad_kernel<<<4096, 256, 0, stream>>>(wq, wqb, (long)NL * NE, NE, NE);
  convert_pad_kernel<<<4096, 256, 0, stream>>>(wk, wkb, (long)NL * NE, NE, NE);
  convert_pad_kernel<<<4096, 256, 0, stream>>>(wv, wvb, (long)NL * NE, NE, NE);
  convert_pad_kernel<<<4096, 256, 0, stream>>>(wo, wob, (long)NL * NE, NE, NE);
  convert_pad_kernel<<<4096, 256, 0, stream>>>(w1, w1b, (long)NL * NE, NHID, NHID);
  convert_pad_kernel<<<4096, 256, 0, stream>>>(w2, w2b, (long)NL * NHID, NE, NE);
  convert_pad_kernel<<<4096, 256, 0, stream>>>(fc1_w, fc1b, NE, NE, NE);
  convert_pad_kernel<<<4096, 256, 0, stream>>>(fc2_w, fc2b, NE, NOUT, NOUTP);

  // ---- x0 = LN(embed), attention biases ----
  embed_ln_kernel<<<NM, 256, 0, stream>>>(src, ele_emb, x0f, x0b);
  bias_build_kernel<<<(NB * NH * NT + 255) / 256, 256, 0, stream>>>(
      dist, rel, mask, sp_emb, graph_virt, rel_emb, rel_w, bias0, bias1, bias2);

  const float* encBias[3] = {bias0, bias1, bias2};

  for (int enc = 0; enc < 3; enc++) {
    const float* curxf = x0f;
    const unsigned short* curxb = x0b;
    for (int l = 0; l < NL; l++) {
      const unsigned short* wq_l = wqb + (size_t)l * NE * NE;
      const unsigned short* wk_l = wkb + (size_t)l * NE * NE;
      const unsigned short* wv_l = wvb + (size_t)l * NE * NE;
      const unsigned short* wo_l = wob + (size_t)l * NE * NE;
      const unsigned short* w1_l = w1b + (size_t)l * NE * NHID;
      const unsigned short* w2_l = w2b + (size_t)l * NHID * NE;

      launch_gemm(stream, curxb, wq_l, bq + l * NE, qf, NM, NE, NE, NE, 0);
      launch_gemm(stream, curxb, wk_l, bk + l * NE, kf, NM, NE, NE, NE, 0);
      launch_gemm(stream, curxb, wv_l, bv + l * NE, vf, NM, NE, NE, NE, 0);
      attn_kernel<<<(NB * NT * NH) / 8, 256, 0, stream>>>(qf, kf, vf, encBias[enc], ctxb);
      launch_gemm(stream, ctxb, wo_l, bo + l * NE, qf, NM, NE, NE, NE, 0);
      residual_ln_kernel<<<NM, 256, 0, stream>>>(curxf, qf, ln1g + l * NE, ln1b + l * NE, xf, xb);
      curxf = xf; curxb = xb;
      launch_gemm(stream, xb, w1_l, b1 + (size_t)l * NHID, h1b, NM, NE, NHID, NHID, 3); // gelu->bf16
      launch_gemm(stream, h1b, w2_l, b2 + l * NE, qf, NM, NHID, NE, NE, 0);
      residual_ln_kernel<<<NM, 256, 0, stream>>>(xf, qf, ln2g + l * NE, ln2b + l * NE, xf, xb);
    }
    // head
    const int tok = (enc == 0) ? 0 : 11;
    gather_rows_kernel<<<(NB * NE) / 256, 256, 0, stream>>>(xb, selb, tok);
    launch_gemm(stream, selb, fc1b, fc1_b, g1f, NB, NE, NE, NE, 0);
    head_mid_kernel<<<NB, 256, 0, stream>>>(g1f, post_scale, post_bias, tb);
    float* lgout = (enc == 0) ? (float*)d_out : lgt;
    launch_gemm(stream, tb, fc2b, fc2_b, lgout, NB, NE, NOUTP, NOUT, 0);
    if (enc == 1) loss_row_kernel<<<NB, 256, 0, stream>>>(lgt, target, lb1);
    if (enc == 2) loss_row_kernel<<<NB, 256, 0, stream>>>(lgt, target, lb2);
  }
  loss_final_kernel<<<1, 256, 0, stream>>>(lb1, lb2, (float*)d_out + (size_t)NB * NOUT);
}
